// xLSTMNet_37821482008977
// MI455X (gfx1250) — matrix operand order repaired and checked by reading, not compiled, not run on hardware
//
#include <hip/hip_runtime.h>
#include <cstdint>
#include <cstddef>

// ---------------- problem constants (match reference) ----------------
#define B_      8
#define S_      1024
#define E_      512
#define INNER_  1024
#define NH_     4
#define DH_     256
#define KS_     4
#define NC_     1000
#define NL_     4

typedef __bf16 bf16;
typedef __attribute__((ext_vector_type(16))) __bf16 v16bf;
typedef __attribute__((ext_vector_type(8)))  __bf16 v8bf;
typedef __attribute__((ext_vector_type(8)))  float  v8f;

__device__ inline v8bf ldv8(const bf16* p) { return *(const v8bf*)p; }
__device__ inline v16bf cat16(v8bf lo, v8bf hi) {
    v16bf r;
#pragma unroll
    for (int i = 0; i < 8; ++i) { r[i] = lo[i]; r[i + 8] = hi[i]; }
    return r;
}
__device__ inline v8f zero8f() {
    v8f v;
#pragma unroll
    for (int i = 0; i < 8; ++i) v[i] = 0.f;
    return v;
}
__device__ inline v8bf zero8b() {
    v8bf v;
#pragma unroll
    for (int i = 0; i < 8; ++i) v[i] = (bf16)0.f;
    return v;
}
__device__ inline float siluf(float x) { return x / (1.f + expf(-x)); }

// ---- CDNA5 async global->LDS copy (ASYNCcnt path), per ISA 08_async_tensor ----
// LDS generic pointers on AMDGPU carry the LDS byte address in addr[31:0].
__device__ inline void async_b128(uint32_t lds_addr, const void* gsrc) {
    asm volatile("global_load_async_to_lds_b128 %0, %1, off"
                 :: "v"(lds_addr), "v"(gsrc)
                 : "memory");
}
__device__ inline void wait_async0() {
    asm volatile("s_wait_asynccnt 0x0" ::: "memory");
}
__device__ inline uint32_t lds_addr32(const void* p) {
    return (uint32_t)(uintptr_t)p;
}

// ---------------- embedding gather ----------------
__global__ void k_embed(const int* __restrict__ tok, const float* __restrict__ emb,
                        float* __restrict__ h, int n) {
    int i = blockIdx.x * blockDim.x + threadIdx.x;
    if (i >= n) return;
    int row = i / E_;
    int e = i - row * E_;
    h[i] = emb[(size_t)tok[row] * E_ + e];
}

// ---------------- f32 -> bf16 convert ----------------
__global__ void k_cvt(const float* __restrict__ in, bf16* __restrict__ out, size_t n) {
    size_t i = (size_t)blockIdx.x * blockDim.x + threadIdx.x;
    if (i < n) out[i] = (bf16)in[i];
}

// ---------------- weight-only LayerNorm -> bf16 ----------------
__global__ __launch_bounds__(256) void k_layernorm_bf16(
    const float* __restrict__ x, size_t row_stride, const float* __restrict__ w,
    bf16* __restrict__ y, int C) {
    __shared__ float rs[8], rq[8];
    const int tid = threadIdx.x, lane = tid & 31, wave = tid >> 5;
    const float* xr = x + (size_t)blockIdx.x * row_stride;
    float s = 0.f, sq = 0.f;
    for (int i = tid; i < C; i += 256) { float v = xr[i]; s += v; sq += v * v; }
#pragma unroll
    for (int off = 16; off >= 1; off >>= 1) {
        s += __shfl_xor(s, off, 32);
        sq += __shfl_xor(sq, off, 32);
    }
    if (lane == 0) { rs[wave] = s; rq[wave] = sq; }
    __syncthreads();
    if (tid == 0) {
        float a = 0.f, b = 0.f;
        for (int i = 0; i < 8; ++i) { a += rs[i]; b += rq[i]; }
        rs[0] = a; rq[0] = b;
    }
    __syncthreads();
    float mean = rs[0] / (float)C;
    float var = rq[0] / (float)C - mean * mean;
    float rstd = rsqrtf(var + 1e-5f);
    bf16* yr = y + (size_t)blockIdx.x * C;
    for (int i = tid; i < C; i += 256) yr[i] = (bf16)((xr[i] - mean) * rstd * w[i]);
}

// ---------------- generic bf16 WMMA GEMM: out = A(MxK) * W(NxK)^T + bias (+res) ----------------
// 256 threads = 8 waves; block tile 64(M) x 128(N); K stepped by 32.
// Full tiles: double-buffered LDS fed by GLOBAL_LOAD_ASYNC_TO_LDS_B128 (ASYNCcnt).
__global__ __launch_bounds__(256) void k_gemm_bf16(
    const bf16* __restrict__ A, const bf16* __restrict__ W,
    const float* __restrict__ bias, const float* __restrict__ res,
    float* __restrict__ out, int M, int N, int K) {
    __shared__ __align__(32) bf16 sA[2][64 * 32];
    __shared__ __align__(32) bf16 sB[2][128 * 32];
    const int tid = threadIdx.x;
    const int lane = tid & 31;
    const int wave = tid >> 5;
    const int mt = wave >> 1;      // 0..3 : M sub-tile
    const int ng = wave & 1;       // 0..1 : N half (64 cols)
    const int ml = lane & 15;
    const int hk = lane >> 4;      // lane half selects K sub-chunks

    const int m0 = blockIdx.y * 64;
    const int n0 = blockIdx.x * 128;
    const bool full = (m0 + 64 <= M) && (n0 + 128 <= N);

    v8f acc[4];
#pragma unroll
    for (int j = 0; j < 4; ++j) acc[j] = zero8f();

    const int ra = tid >> 2;          // 0..63
    const int ca = (tid & 3) * 8;     // 0/8/16/24
    const int nK = K >> 5;

    if (full) {
        {   // prologue: async-stage tile 0 into buffer 0
            async_b128(lds_addr32(&sA[0][ra * 32 + ca]),
                       A + (size_t)(m0 + ra) * K + ca);
            async_b128(lds_addr32(&sB[0][ra * 32 + ca]),
                       W + (size_t)(n0 + ra) * K + ca);
            async_b128(lds_addr32(&sB[0][(ra + 64) * 32 + ca]),
                       W + (size_t)(n0 + ra + 64) * K + ca);
        }
        for (int ks = 0; ks < nK; ++ks) {
            wait_async0();
            __syncthreads();
            const int cur = ks & 1;
            if (ks + 1 < nK) {   // stream next tile into the other buffer
                const int kn = (ks + 1) << 5;
                const int nxt = cur ^ 1;
                async_b128(lds_addr32(&sA[nxt][ra * 32 + ca]),
                           A + (size_t)(m0 + ra) * K + kn + ca);
                async_b128(lds_addr32(&sB[nxt][ra * 32 + ca]),
                           W + (size_t)(n0 + ra) * K + kn + ca);
                async_b128(lds_addr32(&sB[nxt][(ra + 64) * 32 + ca]),
                           W + (size_t)(n0 + ra + 64) * K + kn + ca);
            }
            const bf16* ap = &sA[cur][(mt * 16 + ml) * 32];
            v16bf afrag = cat16(ldv8(ap + hk * 16), ldv8(ap + hk * 16 + 8));
            const bf16* bb = &sB[cur][(ng * 64 + ml) * 32 + hk * 16];
            v16bf b0 = *(const v16bf*)(bb + 0 * 16 * 32);
            v16bf b1 = *(const v16bf*)(bb + 1 * 16 * 32);
            v16bf b2 = *(const v16bf*)(bb + 2 * 16 * 32);
            v16bf b3 = *(const v16bf*)(bb + 3 * 16 * 32);
            acc[0] = __builtin_amdgcn_wmma_f32_16x16x32_bf16(false, afrag, false, b0, (short)0, acc[0], false, false);
            acc[1] = __builtin_amdgcn_wmma_f32_16x16x32_bf16(false, afrag, false, b1, (short)0, acc[1], false, false);
            acc[2] = __builtin_amdgcn_wmma_f32_16x16x32_bf16(false, afrag, false, b2, (short)0, acc[2], false, false);
            acc[3] = __builtin_amdgcn_wmma_f32_16x16x32_bf16(false, afrag, false, b3, (short)0, acc[3], false, false);
        }
    } else {
        // guarded synchronous path for edge tiles (classifier GEMM)
        for (int ks = 0; ks < nK; ++ks) {
            const int k0 = ks << 5;
            __syncthreads();
            {
                int gm = m0 + ra;
                v8bf v = zero8b();
                if (gm < M) v = ldv8(A + (size_t)gm * K + k0 + ca);
                *(v8bf*)&sA[0][ra * 32 + ca] = v;
            }
#pragma unroll
            for (int hh = 0; hh < 2; ++hh) {
                int rb = ra + hh * 64;
                int gn = n0 + rb;
                v8bf v = zero8b();
                if (gn < N) v = ldv8(W + (size_t)gn * K + k0 + ca);
                *(v8bf*)&sB[0][rb * 32 + ca] = v;
            }
            __syncthreads();
            const bf16* ap = &sA[0][(mt * 16 + ml) * 32];
            v16bf afrag = cat16(ldv8(ap + hk * 16), ldv8(ap + hk * 16 + 8));
            const bf16* bb = &sB[0][(ng * 64 + ml) * 32 + hk * 16];
            v16bf b0 = *(const v16bf*)(bb + 0 * 16 * 32);
            v16bf b1 = *(const v16bf*)(bb + 1 * 16 * 32);
            v16bf b2 = *(const v16bf*)(bb + 2 * 16 * 32);
            v16bf b3 = *(const v16bf*)(bb + 3 * 16 * 32);
            acc[0] = __builtin_amdgcn_wmma_f32_16x16x32_bf16(false, afrag, false, b0, (short)0, acc[0], false, false);
            acc[1] = __builtin_amdgcn_wmma_f32_16x16x32_bf16(false, afrag, false, b1, (short)0, acc[1], false, false);
            acc[2] = __builtin_amdgcn_wmma_f32_16x16x32_bf16(false, afrag, false, b2, (short)0, acc[2], false, false);
            acc[3] = __builtin_amdgcn_wmma_f32_16x16x32_bf16(false, afrag, false, b3, (short)0, acc[3], false, false);
        }
    }
    // epilogue: C/D layout -> lanes 0-15 rows r, lanes 16-31 rows r+8
#pragma unroll
    for (int j = 0; j < 4; ++j) {
        int gn = n0 + ng * 64 + j * 16 + ml;
#pragma unroll
        for (int r = 0; r < 8; ++r) {
            int gm = m0 + mt * 16 + r + hk * 8;
            if (gm < M && gn < N) {
                float v = acc[j][r];
                if (bias) v += bias[gn];
                if (res)  v += res[(size_t)gm * N + gn];
                out[(size_t)gm * N + gn] = v;
            }
        }
    }
}

// ---------------- causal depthwise conv (KS=4) + SiLU ----------------
__global__ void k_conv_silu(const float* __restrict__ xz, const float* __restrict__ w,
                            const float* __restrict__ bias, float* __restrict__ xc, int n) {
    int i = blockIdx.x * blockDim.x + threadIdx.x;
    if (i >= n) return;
    int c = i % INNER_;
    int row = i / INNER_;       // b*S + s
    int s = row % S_;
    int b = row / S_;
    float acc = bias[c];
#pragma unroll
    for (int k = 0; k < KS_; ++k) {
        int ds = s - (KS_ - 1) + k;
        if (ds >= 0)
            acc += w[c * KS_ + k] * xz[((size_t)b * S_ + ds) * (2 * INNER_) + c];
    }
    xc[i] = siluf(acc);
}

// ---------------- blockwise 4x4 headwise linear (f32 + bf16 outputs) ----------------
__global__ void k_headwise(const float* __restrict__ xin, size_t row_stride,
                           const float* __restrict__ w, const float* __restrict__ bias,
                           float* __restrict__ yf, bf16* __restrict__ yb, int n) {
    int i = blockIdx.x * blockDim.x + threadIdx.x;
    if (i >= n) return;
    int c = i % INNER_;
    int row = i / INNER_;
    int hh = c >> 2, o = c & 3;
    const float* xr = xin + (size_t)row * row_stride + (hh << 2);
    const float* wr = w + hh * 16 + o * 4;
    float v = bias[c] + xr[0] * wr[0] + xr[1] * wr[1] + xr[2] * wr[2] + xr[3] * wr[3];
    yf[i] = v;
    yb[i] = (bf16)v;
}

// ---------------- gate preactivations: igp/fgp = [q,k,v] . w + b ----------------
__global__ __launch_bounds__(256) void k_gates(
    const float* __restrict__ qf, const float* __restrict__ kf, const float* __restrict__ vf,
    const float* __restrict__ ig_w, const float* __restrict__ ig_b,
    const float* __restrict__ fg_w, const float* __restrict__ fg_b,
    float* __restrict__ igp, float* __restrict__ fgp) {
    int row = blockIdx.x;   // b*S + s
    int b = row / S_, s = row % S_;
    int tid = threadIdx.x, lane = tid & 31, wave = tid >> 5;
    int hh = wave & 3;
    bool isf = wave >= 4;
    const float* wv = (isf ? fg_w : ig_w) + (size_t)hh * (3 * INNER_);
    const float* qr = qf + (size_t)row * INNER_;
    const float* kr = kf + (size_t)row * INNER_;
    const float* vr = vf + (size_t)row * INNER_;
    float acc = 0.f;
    for (int j = lane; j < INNER_; j += 32)
        acc += qr[j] * wv[j] + kr[j] * wv[INNER_ + j] + vr[j] * wv[2 * INNER_ + j];
#pragma unroll
    for (int off = 16; off >= 1; off >>= 1) acc += __shfl_xor(acc, off, 32);
    if (lane == 0) {
        float r = acc + (isf ? fg_b[hh] : ig_b[hh]);
        float* dst = isf ? fgp : igp;
        dst[((size_t)b * NH_ + hh) * S_ + s] = r;
    }
}

// ---------------- lfc = pad(cumsum(log_sigmoid(fgp))) per (b,nh), wave scan ----------------
__global__ __launch_bounds__(32) void k_cumsum(const float* __restrict__ fgp,
                                               float* __restrict__ lfc) {
    int seq = blockIdx.x;            // b*NH + nh
    int lane = threadIdx.x;
    const float* f = fgp + (size_t)seq * S_;
    float* o = lfc + (size_t)seq * (S_ + 1);
    if (lane == 0) o[0] = 0.f;
    float carry = 0.f;
    for (int c = 0; c < S_ / 32; ++c) {
        float x = f[c * 32 + lane];
        float v = (x >= 0.f) ? -log1pf(expf(-x)) : (x - log1pf(expf(x)));
#pragma unroll
        for (int d = 1; d < 32; d <<= 1) {
            float t = __shfl_up(v, d, 32);
            if (lane >= d) v += t;
        }
        float tot = __shfl(v, 31, 32);
        v += carry;
        carry += tot;
        o[c * 32 + lane + 1] = v;
    }
}

// ---------------- flash-style mLSTM cell + headwise group-norm ----------------
// grid = B*NH*(S/64), block = 128 (4 waves, each owning a 16-row query tile).
__global__ __launch_bounds__(128) void k_mlstm_flash(
    const bf16* __restrict__ qb, const bf16* __restrict__ kb, const bf16* __restrict__ vb,
    const float* __restrict__ igp, const float* __restrict__ lfc,
    const float* __restrict__ on_w, float* __restrict__ hcell) {
    __shared__ __align__(32) bf16 kt[32 * DH_];     // (t, d)
    __shared__ __align__(32) bf16 vt[DH_ * 32];     // (d, t) transposed
    __shared__ __align__(32) bf16 pt[4][16 * 32];   // per-wave P tile (s, t)

    const int tid = threadIdx.x, lane = tid & 31, wave = tid >> 5;
    const int ml = lane & 15, hk = lane >> 4;

    const int nblk = S_ / 64;
    int wg = blockIdx.x;
    int sblk = wg % nblk;
    int bn = wg / nblk;
    int nh = bn % NH_;
    int b = bn / NH_;

    const int s_base = sblk * 64 + wave * 16;
    const bf16* qh = qb + (size_t)b * S_ * INNER_ + nh * DH_;
    const bf16* kh = kb + (size_t)b * S_ * INNER_ + nh * DH_;
    const bf16* vh = vb + (size_t)b * S_ * INNER_ + nh * DH_;
    const float* ig = igp + (size_t)(b * NH_ + nh) * S_;
    const float* lf = lfc + (size_t)(b * NH_ + nh) * (S_ + 1) + 1;  // lf[t] = lfc[t+1]

    // preload Q fragments over full DH (8 chunks of K=32)
    v16bf qf[8];
#pragma unroll
    for (int c = 0; c < 8; ++c) {
        const bf16* qp = qh + (size_t)(s_base + ml) * INNER_ + c * 32 + hk * 16;
        qf[c] = cat16(ldv8(qp), ldv8(qp + 8));
    }
    float lfcs[8];
#pragma unroll
    for (int r = 0; r < 8; ++r) lfcs[r] = lf[s_base + r + hk * 8];

    float mrun[8], lrun[8];
#pragma unroll
    for (int r = 0; r < 8; ++r) { mrun[r] = -1e30f; lrun[r] = 0.f; }
    v8f acc[16];
#pragma unroll
    for (int c = 0; c < 16; ++c) acc[c] = zero8f();

    const float inv_sqrt_dh = 0.0625f;   // 1/sqrt(256)
    const int nT = sblk * 2 + 2;         // 32-key tiles covering [0, (sblk+1)*64)

    for (int tt = 0; tt < nT; ++tt) {
        const int t0 = tt * 32;
        __syncthreads();
        {   // stage K row-major via async-to-LDS; V^T via VGPR transpose
            int tr = tid >> 2;               // 0..31
            int c0 = (tid & 3) * 64;         // 0/64/128/192
            const bf16* ksrc = kh + (size_t)(t0 + tr) * INNER_ + c0;
            uint32_t kdst = lds_addr32(&kt[tr * DH_ + c0]);
#pragma unroll
            for (int u = 0; u < 8; ++u) async_b128(kdst + u * 16, ksrc + u * 8);
            const bf16* vsrc = vh + (size_t)(t0 + tr) * INNER_ + c0;
#pragma unroll
            for (int u = 0; u < 8; ++u) {
                v8bf vv = ldv8(vsrc + u * 8);
#pragma unroll
                for (int e = 0; e < 8; ++e) vt[(c0 + u * 8 + e) * 32 + tr] = vv[e];
            }
            wait_async0();
        }
        __syncthreads();
        if (t0 > s_base + 15) continue;   // wave fully causal-masked; barriers stay matched

        // QK^T scores: two independent 16-key accumulator chains sharing Q frags
        v8f sc0 = zero8f(), sc1 = zero8f();
#pragma unroll
        for (int c = 0; c < 8; ++c) {
            v16bf bk0 = *(const v16bf*)&kt[(ml) * DH_ + c * 32 + hk * 16];
            v16bf bk1 = *(const v16bf*)&kt[(16 + ml) * DH_ + c * 32 + hk * 16];
            sc0 = __builtin_amdgcn_wmma_f32_16x16x32_bf16(false, qf[c], false, bk0, (short)0, sc0, false, false);
            sc1 = __builtin_amdgcn_wmma_f32_16x16x32_bf16(false, qf[c], false, bk1, (short)0, sc1, false, false);
        }
        // decay gating + online max rescale; normalizer accumulates signed sum(C)
        float acol0, acol1;
        {
            int tg0 = t0 + ml, tg1 = t0 + 16 + ml;
            acol0 = ig[tg0] - lf[tg0];
            acol1 = ig[tg1] - lf[tg1];
        }
#pragma unroll
        for (int r = 0; r < 8; ++r) {
            const int srow = s_base + r + hk * 8;
            float ld0 = (t0 + ml <= srow)      ? (lfcs[r] + acol0) : -1e30f;
            float ld1 = (t0 + 16 + ml <= srow) ? (lfcs[r] + acol1) : -1e30f;
            float tmax = fmaxf(ld0, ld1);
#pragma unroll
            for (int off = 1; off < 16; off <<= 1)
                tmax = fmaxf(tmax, __shfl_xor(tmax, off, 32));
            float mnew = fmaxf(mrun[r], tmax);
            float fac = expf(mrun[r] - mnew);
            mrun[r] = mnew;
            float p0 = (ld0 > -1e29f) ? sc0[r] * inv_sqrt_dh * expf(ld0 - mnew) : 0.f;
            float p1 = (ld1 > -1e29f) ? sc1[r] * inv_sqrt_dh * expf(ld1 - mnew) : 0.f;
            float psum = p0 + p1;
#pragma unroll
            for (int off = 1; off < 16; off <<= 1)
                psum += __shfl_xor(psum, off, 32);
            lrun[r] = lrun[r] * fac + psum;
            pt[wave][(r + hk * 8) * 32 + ml] = (bf16)p0;
            pt[wave][(r + hk * 8) * 32 + 16 + ml] = (bf16)p1;
#pragma unroll
            for (int c = 0; c < 16; ++c) acc[c][r] *= fac;
        }
        // PV: P (16x32) from per-wave LDS as A-frag, V^T as B-frags; two chains
        const bf16* pp = &pt[wave][ml * 32];
        v16bf pfrag = cat16(ldv8(pp + hk * 16), ldv8(pp + hk * 16 + 8));
#pragma unroll
        for (int c = 0; c < 16; c += 2) {
            v16bf bv0 = *(const v16bf*)&vt[(c * 16 + ml) * 32 + hk * 16];
            v16bf bv1 = *(const v16bf*)&vt[((c + 1) * 16 + ml) * 32 + hk * 16];
            acc[c]     = __builtin_amdgcn_wmma_f32_16x16x32_bf16(false, pfrag, false, bv0, (short)0, acc[c], false, false);
            acc[c + 1] = __builtin_amdgcn_wmma_f32_16x16x32_bf16(false, pfrag, false, bv1, (short)0, acc[c + 1], false, false);
        }
    }

    // epilogue: xLSTM normalizer, then weight-only group-norm over DH
#pragma unroll
    for (int r = 0; r < 8; ++r) {
        float nrm = fmaxf(fabsf(lrun[r]), expf(-mrun[r])) + 1e-6f;
        float rinv = 1.f / nrm;
        float sum = 0.f, sq = 0.f;
#pragma unroll
        for (int c = 0; c < 16; ++c) {
            float x = acc[c][r] * rinv;
            acc[c][r] = x;
            sum += x;
            sq += x * x;
        }
#pragma unroll
        for (int off = 1; off < 16; off <<= 1) {
            sum += __shfl_xor(sum, off, 32);
            sq += __shfl_xor(sq, off, 32);
        }
        float mu = sum * (1.f / DH_);
        float var = sq * (1.f / DH_) - mu * mu;
        float rstd = rsqrtf(var + 1e-5f);
        int srow = s_base + r + hk * 8;
        float* orow = hcell + ((size_t)b * S_ + srow) * INNER_ + nh * DH_;
#pragma unroll
        for (int c = 0; c < 16; ++c) {
            int d = c * 16 + ml;
            orow[d] = (acc[c][r] - mu) * rstd * on_w[nh * DH_ + d];
        }
    }
}

// ---------------- h2 = (hcell + skip*xc) * silu(z)  -> bf16 ----------------
__global__ void k_mix(const float* __restrict__ hcell, const float* __restrict__ xc,
                      const float* __restrict__ skip, const float* __restrict__ xz,
                      bf16* __restrict__ hmixb, int n) {
    int i = blockIdx.x * blockDim.x + threadIdx.x;
    if (i >= n) return;
    int c = i % INNER_;
    int row = i / INNER_;
    float z = xz[(size_t)row * (2 * INNER_) + INNER_ + c];
    float v = (hcell[i] + skip[c] * xc[i]) * siluf(z);
    hmixb[i] = (bf16)v;
}

// ==================== host orchestration ====================
extern "C" void kernel_launch(void* const* d_in, const int* in_sizes, int n_in,
                              void* d_out, int out_size, void* d_ws, size_t ws_size,
                              hipStream_t stream) {
    (void)in_sizes; (void)n_in; (void)out_size; (void)ws_size;

    const int* x = (const int*)d_in[0];
    const float* emb = (const float*)d_in[1];
    auto blk = [&](int L, int k) -> const float* { return (const float*)d_in[2 + L * 19 + k]; };
    const float* post_ln_w = (const float*)d_in[2 + NL_ * 19 + 0];
    const float* cls_w     = (const float*)d_in[2 + NL_ * 19 + 1];
    const float* cls_b     = (const float*)d_in[2 + NL_ * 19 + 2];

    char* wsb = (char*)d_ws;
    size_t off = 0;
    auto ALLOC = [&](size_t bytes) -> void* {
        void* p = (void*)(wsb + off);
        off += (bytes + 255) & ~(size_t)255;
        return p;
    };
    const size_t BS = (size_t)B_ * S_;
    float* h     = (float*)ALLOC(BS * E_ * 4);
    bf16*  xnb   = (bf16*)ALLOC(BS * E_ * 2);
    float* xz    = (float*)ALLOC(BS * 2 * INNER_ * 4);
    float* xc    = (float*)ALLOC(BS * INNER_ * 4);
    float* qf    = (float*)ALLOC(BS * INNER_ * 4);
    float* kf    = (float*)ALLOC(BS * INNER_ * 4);
    float* vf    = (float*)ALLOC(BS * INNER_ * 4);
    bf16*  qb    = (bf16*)ALLOC(BS * INNER_ * 2);
    bf16*  kb    = (bf16*)ALLOC(BS * INNER_ * 2);
    bf16*  vb    = (bf16*)ALLOC(BS * INNER_ * 2);
    float* hcell = (float*)ALLOC(BS * INNER_ * 4);
    float* igp   = (float*)ALLOC((size_t)B_ * NH_ * S_ * 4);
    float* fgp   = (float*)ALLOC((size_t)B_ * NH_ * S_ * 4);
    float* lfc   = (float*)ALLOC((size_t)B_ * NH_ * (S_ + 1) * 4);
    bf16*  upwb  = (bf16*)ALLOC((size_t)2 * INNER_ * E_ * 2);
    bf16*  dwwb  = (bf16*)ALLOC((size_t)E_ * INNER_ * 2);
    bf16*  clswb = (bf16*)ALLOC((size_t)NC_ * E_ * 2);
    bf16*  lastb = (bf16*)ALLOC((size_t)B_ * E_ * 2);
    bf16*  hmixb = qb;   // safe reuse: cell consumed qb before k_mix runs

    const int nEmb = (int)(BS * E_);
    const int nInner = (int)(BS * INNER_);

    k_embed<<<(nEmb + 255) / 256, 256, 0, stream>>>(x, emb, h, nEmb);

    for (int L = 0; L < NL_; ++L) {
        const float* ln_w   = blk(L, 0);
        const float* up_w   = blk(L, 1);
        const float* up_b   = blk(L, 2);
        const float* conv_w = blk(L, 3);
        const float* conv_b = blk(L, 4);
        const float* q_w    = blk(L, 5);
        const float* q_b    = blk(L, 6);
        const float* k_w    = blk(L, 7);
        const float* k_b    = blk(L, 8);
        const float* v_w    = blk(L, 9);
        const float* v_b    = blk(L, 10);
        const float* ig_w   = blk(L, 11);
        const float* ig_b   = blk(L, 12);
        const float* fg_w   = blk(L, 13);
        const float* fg_b   = blk(L, 14);
        const float* on_w   = blk(L, 15);
        const float* skip   = blk(L, 16);
        const float* down_w = blk(L, 17);
        const float* down_b = blk(L, 18);

        size_t nupw = (size_t)2 * INNER_ * E_;
        size_t ndww = (size_t)E_ * INNER_;
        k_cvt<<<(unsigned)((nupw + 255) / 256), 256, 0, stream>>>(up_w, upwb, nupw);
        k_cvt<<<(unsigned)((ndww + 255) / 256), 256, 0, stream>>>(down_w, dwwb, ndww);

        k_layernorm_bf16<<<(unsigned)BS, 256, 0, stream>>>(h, (size_t)E_, ln_w, xnb, E_);

        {   // up-projection: M=8192, N=2048, K=512
            dim3 g((2 * INNER_ + 127) / 128, (unsigned)((BS + 63) / 64));
            k_gemm_bf16<<<g, 256, 0, stream>>>(xnb, upwb, up_b, nullptr, xz,
                                               (int)BS, 2 * INNER_, E_);
        }
        k_conv_silu<<<(nInner + 255) / 256, 256, 0, stream>>>(xz, conv_w, conv_b, xc, nInner);

        k_headwise<<<(nInner + 255) / 256, 256, 0, stream>>>(xc, (size_t)INNER_, q_w, q_b, qf, qb, nInner);
        k_headwise<<<(nInner + 255) / 256, 256, 0, stream>>>(xc, (size_t)INNER_, k_w, k_b, kf, kb, nInner);
        k_headwise<<<(nInner + 255) / 256, 256, 0, stream>>>(xz, (size_t)(2 * INNER_), v_w, v_b, vf, vb, nInner);

        k_gates<<<(unsigned)BS, 256, 0, stream>>>(qf, kf, vf, ig_w, ig_b, fg_w, fg_b, igp, fgp);
        k_cumsum<<<B_ * NH_, 32, 0, stream>>>(fgp, lfc);

        k_mlstm_flash<<<B_ * NH_ * (S_ / 64), 128, 0, stream>>>(qb, kb, vb, igp, lfc, on_w, hcell);

        k_mix<<<(nInner + 255) / 256, 256, 0, stream>>>(hcell, xc, skip, xz, hmixb, nInner);

        {   // down-projection with residual into h: M=8192, N=512, K=1024
            dim3 g((E_ + 127) / 128, (unsigned)((BS + 63) / 64));
            k_gemm_bf16<<<g, 256, 0, stream>>>(hmixb, dwwb, down_b, h, h,
                                               (int)BS, E_, INNER_);
        }
    }

    // final: post-LN on last token, classifier GEMM (M=8 padded in-kernel, N=1000)
    size_t nclsw = (size_t)NC_ * E_;
    k_cvt<<<(unsigned)((nclsw + 255) / 256), 256, 0, stream>>>(cls_w, clswb, nclsw);
    k_layernorm_bf16<<<B_, 256, 0, stream>>>(h + (size_t)(S_ - 1) * E_, (size_t)S_ * E_,
                                             post_ln_w, lastb, E_);
    {
        dim3 g((NC_ + 127) / 128, 1);
        k_gemm_bf16<<<g, 256, 0, stream>>>(lastb, clswb, cls_b, nullptr,
                                           (float*)d_out, B_, NC_, E_);
    }
}
